// CrystalDiffusionModel_27092653703529
// MI455X (gfx1250) — matrix operand order repaired and checked by reading, not compiled, not run on hardware
//
#include <hip/hip_runtime.h>
#include <hip/hip_bf16.h>
#include <math.h>

#define NN 50000
#define NE 800000
#define HID 128
#define NG 8

typedef _Float16 v16h __attribute__((ext_vector_type(16)));
typedef _Float16 h8   __attribute__((ext_vector_type(8)));
typedef float    v8f  __attribute__((ext_vector_type(8)));

__device__ __forceinline__ float silu_f(float x) { return x / (1.f + __expf(-x)); }

// Assemble a 16x32 f16 A-fragment for one lane.
// Contiguous order: lanes 0-15 (khalf=0) hold K = k0+{0..15};
// lanes 16-31 (khalf=1) hold K = k0+{16..31}.
__device__ __forceinline__ v16h ld_afrag16(const _Float16* base, int khalf) {
  h8 lo = *(const h8*)(base + khalf * 16);
  h8 hi = *(const h8*)(base + khalf * 16 + 8);
  v16h r;
#pragma unroll
  for (int j = 0; j < 8; ++j) { r[j] = lo[j]; r[j + 8] = hi[j]; }
  return r;
}

#define WMMA_F16(a, b, c) \
  __builtin_amdgcn_wmma_f32_16x16x32_f16(false, (a), false, (b), (short)0, (c), false, false)

// ---------------------------------------------------------------------------
// Kernel 0: all tiny per-graph math (time embedding MLP, cond MLPs, combined
// cond MLP, expanded ctx, K/V projections of ctx). One block, 256 threads.
// ---------------------------------------------------------------------------
__global__ void precompute_kernel(
    const float* t, const float* topo, const float* stab, const float* sust,
    const float* tW1, const float* tb1, const float* tW2, const float* tb2,
    const float* toW1, const float* tob1, const float* toW2, const float* tob2,
    const float* stW1, const float* stb1, const float* stW2, const float* stb2,
    const float* suW1, const float* sub1, const float* suW2, const float* sub2,
    const float* cW1, const float* cb1, const float* cW2, const float* cb2,
    const float* Wk, const float* Wv,
    float* tinf, float* kctx, float* vctx) {
  __shared__ float sEmb[NG * 128];
  __shared__ float sHid[NG * 256];
  __shared__ float sCat[NG * 64];
  __shared__ float sTmp[NG * 64];
  __shared__ float sCEmb[NG * 64];
  const int tid = threadIdx.x;

  // sinusoidal time embedding: [8,128]
  for (int idx = tid; idx < NG * 128; idx += 256) {
    int b = idx >> 7, i = idx & 127;
    int hi = i & 63;
    float f = __expf((-logf(10000.f) / 63.f) * (float)hi);
    float a = t[b] * f;
    sEmb[idx] = (i < 64) ? __sinf(a) : __cosf(a);
  }
  __syncthreads();
  // time MLP hidden: [8,256], silu
  for (int idx = tid; idx < NG * 256; idx += 256) {
    int b = idx >> 8, j = idx & 255;
    float acc = tb1[j];
    for (int k = 0; k < 128; ++k) acc += sEmb[b * 128 + k] * tW1[j * 128 + k];
    sHid[idx] = silu_f(acc);
  }
  __syncthreads();
  // time MLP out -> silu -> time_influence [8,128]
  for (int idx = tid; idx < NG * 128; idx += 256) {
    int b = idx >> 7, j = idx & 127;
    float acc = tb2[j];
    for (int k = 0; k < 256; ++k) acc += sHid[b * 256 + k] * tW2[j * 256 + k];
    tinf[idx] = silu_f(acc);
  }
  // cond hiddens (disjoint sTmp ranges)
  for (int idx = tid; idx < NG * 32; idx += 256) {          // topo hidden 8x32
    int b = idx >> 5, j = idx & 31;
    float acc = tob1[j];
    for (int k = 0; k < 7; ++k) acc += topo[b * 7 + k] * toW1[j * 7 + k];
    sTmp[idx] = silu_f(acc);
  }
  for (int idx = tid; idx < NG * 16; idx += 256) {          // stab hidden 8x16
    int b = idx >> 4, j = idx & 15;
    float acc = stb1[j];
    for (int k = 0; k < 2; ++k) acc += stab[b * 2 + k] * stW1[j * 2 + k];
    sTmp[256 + idx] = silu_f(acc);
  }
  for (int idx = tid; idx < NG * 16; idx += 256) {          // sust hidden 8x16
    int b = idx >> 4, j = idx & 15;
    float acc = sub1[j];
    for (int k = 0; k < 3; ++k) acc += sust[b * 3 + k] * suW1[j * 3 + k];
    sTmp[384 + idx] = silu_f(acc);
  }
  __syncthreads();
  // cond outputs -> cond_cat [8,64]
  for (int idx = tid; idx < NG * 32; idx += 256) {
    int b = idx >> 5, j = idx & 31;
    float acc = tob2[j];
    for (int k = 0; k < 32; ++k) acc += sTmp[b * 32 + k] * toW2[j * 32 + k];
    sCat[b * 64 + j] = acc;
  }
  for (int idx = tid; idx < NG * 16; idx += 256) {
    int b = idx >> 4, j = idx & 15;
    float acc = stb2[j];
    for (int k = 0; k < 16; ++k) acc += sTmp[256 + b * 16 + k] * stW2[j * 16 + k];
    sCat[b * 64 + 32 + j] = acc;
  }
  for (int idx = tid; idx < NG * 16; idx += 256) {
    int b = idx >> 4, j = idx & 15;
    float acc = sub2[j];
    for (int k = 0; k < 16; ++k) acc += sTmp[384 + b * 16 + k] * suW2[j * 16 + k];
    sCat[b * 64 + 48 + j] = acc;
  }
  __syncthreads();
  // combined MLP hidden [8,64] silu (reuse sTmp)
  for (int idx = tid; idx < NG * 64; idx += 256) {
    int b = idx >> 6, j = idx & 63;
    float acc = cb1[j];
    for (int k = 0; k < 64; ++k) acc += sCat[b * 64 + k] * cW1[j * 64 + k];
    sTmp[idx] = silu_f(acc);
  }
  __syncthreads();
  // cond_emb [8,64]
  for (int idx = tid; idx < NG * 64; idx += 256) {
    int b = idx >> 6, j = idx & 63;
    float acc = cb2[j];
    for (int k = 0; k < 64; ++k) acc += sTmp[b * 64 + k] * cW2[j * 64 + k];
    sCEmb[idx] = acc;
  }
  __syncthreads();
  // K/V projections of expanded ctx (64 rows; row r = cond_emb[r/8]) -> [64,128]
  for (int idx = tid; idx < 64 * 128; idx += 256) {
    int r = idx >> 7, o = idx & 127, g = r >> 3;
    float ka = 0.f, va = 0.f;
    for (int c = 0; c < 64; ++c) {
      float e = sCEmb[g * 64 + c];
      ka += e * Wk[o * 64 + c];
      va += e * Wv[o * 64 + c];
    }
    kctx[idx] = ka;
    vctx[idx] = va;
  }
}

// ---------------------------------------------------------------------------
// Node / edge encoders (memory bound, tiny K).
// ---------------------------------------------------------------------------
__global__ void node_enc_kernel(const float* x, const float* W, const float* b,
                                float* h, _Float16* hf16) {
  int idx = blockIdx.x * blockDim.x + threadIdx.x;
  if (idx >= NN * HID) return;
  int n = idx >> 7, o = idx & 127;
  float acc = b[o];
#pragma unroll
  for (int k = 0; k < 12; ++k) acc += x[n * 12 + k] * W[o * 12 + k];
  h[idx] = acc;
  hf16[idx] = (_Float16)acc;
}

__global__ void edge_enc_kernel(const float* ea, const float* W, const float* b,
                                _Float16* ef16) {
  long long idx = (long long)blockIdx.x * blockDim.x + threadIdx.x;
  if (idx >= (long long)NE * HID) return;
  int e = (int)(idx >> 7), o = (int)(idx & 127);
  float acc = b[o];
#pragma unroll
  for (int k = 0; k < 20; ++k) acc += ea[e * 20 + k] * W[o * 20 + k];
  ef16[idx] = (_Float16)acc;
}

__global__ void zero_kernel(float* p, int n) {
  int i = blockIdx.x * blockDim.x + threadIdx.x;
  if (i < n) p[i] = 0.f;
}

// ---------------------------------------------------------------------------
// Edge message MLP (WMMA): per 16-edge tile,
//   A[16,384] = [h[src] | h[dst] | e]  (f16)
//   m = silu(A @ W1^T + b1); m = silu(m @ W2^T + b2); atomic scatter to agg[dst].
// Block = 4 waves; weights staged in LDS as f16 once per block.
// ---------------------------------------------------------------------------
__global__ __launch_bounds__(128) void edge_msg_kernel(
    const int* __restrict__ ei, const _Float16* __restrict__ hf16,
    const _Float16* __restrict__ ef16,
    const float* __restrict__ W1, const float* __restrict__ B1,
    const float* __restrict__ W2, const float* __restrict__ B2,
    float* __restrict__ agg, int ntiles, int stride) {
  __shared__ __align__(32) _Float16 sW1[128 * 384];   // 96 KB
  __shared__ __align__(32) _Float16 sW2[128 * 128];   // 32 KB
  __shared__ float sB1[128], sB2[128];
  __shared__ __align__(32) _Float16 sM[4][16 * 128];  // 16 KB, per-wave
  const int tid = threadIdx.x;
  for (int i = tid; i < 128 * 384; i += 128) sW1[i] = (_Float16)W1[i];
  for (int i = tid; i < 128 * 128; i += 128) sW2[i] = (_Float16)W2[i];
  if (tid < 128) { sB1[tid] = B1[tid]; sB2[tid] = B2[tid]; }
  __syncthreads();

  const int wave = tid >> 5, lane = tid & 31;
  const int khalf = lane >> 4, m = lane & 15, ncol = lane & 15;
  _Float16* mrow = &sM[wave][0];

  for (int tile = blockIdx.x * 4 + wave; tile < ntiles; tile += stride) {
    const int edge = tile * 16 + m;
    const int srcn = ei[edge];
    const int dstn = ei[NE + edge];

    v8f acc[8];
    v8f z = {};
#pragma unroll
    for (int nt = 0; nt < 8; ++nt) acc[nt] = z;

    // GEMM1: K = 384 (12 k-tiles of 32)
#pragma unroll
    for (int kt = 0; kt < 12; ++kt) {
      const int k0 = kt * 32;
      const _Float16* ab;
      if (k0 < 128)      ab = hf16 + (size_t)srcn * HID + k0;
      else if (k0 < 256) ab = hf16 + (size_t)dstn * HID + (k0 - 128);
      else               ab = ef16 + (size_t)edge * HID + (k0 - 256);
      v16h af = ld_afrag16(ab, khalf);
#pragma unroll
      for (int nt = 0; nt < 8; ++nt) {
        v16h bf = *(const v16h*)(sW1 + (nt * 16 + ncol) * 384 + k0 + khalf * 16);
        acc[nt] = WMMA_F16(af, bf, acc[nt]);
      }
    }
    // bias + silu, stage m (f16) for transpose via per-wave LDS
#pragma unroll
    for (int nt = 0; nt < 8; ++nt) {
      const int col = nt * 16 + ncol;
      const float bv = sB1[col];
#pragma unroll
      for (int r = 0; r < 8; ++r) {
        float xv = acc[nt][r] + bv;
        xv = silu_f(xv);
        mrow[(r + 8 * khalf) * 128 + col] = (_Float16)xv;
      }
    }
    // GEMM2: K = 128 (4 k-tiles)
#pragma unroll
    for (int nt = 0; nt < 8; ++nt) acc[nt] = z;
#pragma unroll
    for (int kt = 0; kt < 4; ++kt) {
      const int k0 = kt * 32;
      v16h af = ld_afrag16(mrow + m * 128 + k0, khalf);
#pragma unroll
      for (int nt = 0; nt < 8; ++nt) {
        v16h bf = *(const v16h*)(sW2 + (nt * 16 + ncol) * 128 + k0 + khalf * 16);
        acc[nt] = WMMA_F16(af, bf, acc[nt]);
      }
    }
    // bias + silu + atomic scatter-sum into agg[dst]
#pragma unroll
    for (int nt = 0; nt < 8; ++nt) {
      const int col = nt * 16 + ncol;
      const float bv = sB2[col];
#pragma unroll
      for (int r = 0; r < 8; ++r) {
        float xv = silu_f(acc[nt][r] + bv);
        const int row = r + 8 * khalf;
        const int drow = __shfl(dstn, row, 32);
        atomicAdd(agg + (size_t)drow * HID + col, xv);
      }
    }
  }
}

// ---------------------------------------------------------------------------
// Node update MLP (WMMA): u_in = [h | agg] (K=256) -> silu -> K=128 -> hupd.
// ---------------------------------------------------------------------------
__global__ __launch_bounds__(128) void node_upd_kernel(
    const _Float16* __restrict__ hf16, const float* __restrict__ agg,
    const float* __restrict__ W1, const float* __restrict__ B1,
    const float* __restrict__ W2, const float* __restrict__ B2,
    float* __restrict__ hupd, int ntiles, int stride) {
  __shared__ __align__(32) _Float16 sW1[128 * 256];   // 64 KB
  __shared__ __align__(32) _Float16 sW2[128 * 128];   // 32 KB
  __shared__ float sB1[128], sB2[128];
  __shared__ __align__(32) _Float16 sM[4][16 * 128];
  const int tid = threadIdx.x;
  for (int i = tid; i < 128 * 256; i += 128) sW1[i] = (_Float16)W1[i];
  for (int i = tid; i < 128 * 128; i += 128) sW2[i] = (_Float16)W2[i];
  if (tid < 128) { sB1[tid] = B1[tid]; sB2[tid] = B2[tid]; }
  __syncthreads();

  const int wave = tid >> 5, lane = tid & 31;
  const int khalf = lane >> 4, m = lane & 15, ncol = lane & 15;
  _Float16* mrow = &sM[wave][0];

  for (int tile = blockIdx.x * 4 + wave; tile < ntiles; tile += stride) {
    const int node = tile * 16 + m;
    v8f acc[8];
    v8f z = {};
#pragma unroll
    for (int nt = 0; nt < 8; ++nt) acc[nt] = z;

    // GEMM1: K = 256 (first 128 from h f16, next 128 from agg f32->f16)
#pragma unroll
    for (int kt = 0; kt < 8; ++kt) {
      const int k0 = kt * 32;
      v16h af;
      if (k0 < 128) {
        af = ld_afrag16(hf16 + (size_t)node * HID + k0, khalf);
      } else {
        const float* ab = agg + (size_t)node * HID + (k0 - 128);
#pragma unroll
        for (int j = 0; j < 8; ++j) {
          af[j]     = (_Float16)ab[khalf * 16 + j];
          af[j + 8] = (_Float16)ab[khalf * 16 + 8 + j];
        }
      }
#pragma unroll
      for (int nt = 0; nt < 8; ++nt) {
        v16h bf = *(const v16h*)(sW1 + (nt * 16 + ncol) * 256 + k0 + khalf * 16);
        acc[nt] = WMMA_F16(af, bf, acc[nt]);
      }
    }
#pragma unroll
    for (int nt = 0; nt < 8; ++nt) {
      const int col = nt * 16 + ncol;
      const float bv = sB1[col];
#pragma unroll
      for (int r = 0; r < 8; ++r) {
        float xv = silu_f(acc[nt][r] + bv);
        mrow[(r + 8 * khalf) * 128 + col] = (_Float16)xv;
      }
    }
    // GEMM2: K = 128, no activation (outer linear of _mlp2)
#pragma unroll
    for (int nt = 0; nt < 8; ++nt) acc[nt] = z;
#pragma unroll
    for (int kt = 0; kt < 4; ++kt) {
      const int k0 = kt * 32;
      v16h af = ld_afrag16(mrow + m * 128 + k0, khalf);
#pragma unroll
      for (int nt = 0; nt < 8; ++nt) {
        v16h bf = *(const v16h*)(sW2 + (nt * 16 + ncol) * 128 + k0 + khalf * 16);
        acc[nt] = WMMA_F16(af, bf, acc[nt]);
      }
    }
#pragma unroll
    for (int nt = 0; nt < 8; ++nt) {
      const int col = nt * 16 + ncol;
      const float bv = sB2[col];
#pragma unroll
      for (int r = 0; r < 8; ++r) {
        const int row = r + 8 * khalf;
        hupd[(size_t)(tile * 16 + row) * HID + col] = acc[nt][r] + bv;
      }
    }
  }
}

// ---------------------------------------------------------------------------
// LayerNorm + cross-attention (64 ctx tokens, 4 heads x 32) + time + residual.
// One wave per node; per-wave LDS scratch only (no block barriers).
// ---------------------------------------------------------------------------
__global__ __launch_bounds__(128) void ln_attn_kernel(
    const float* __restrict__ hupd, float* __restrict__ hio,
    _Float16* __restrict__ hf16,
    const float* __restrict__ lng, const float* __restrict__ lnb,
    const float* __restrict__ Wq, const float* __restrict__ Wo,
    const float* __restrict__ bo, const float* __restrict__ kctx,
    const float* __restrict__ vctx, const float* __restrict__ tinf,
    const int* __restrict__ batch) {
  __shared__ float sH[4][128];
  __shared__ float sQ[4][128];
  __shared__ float sS[4][256];
  __shared__ float sO[4][128];
  const int wave = threadIdx.x >> 5, lane = threadIdx.x & 31;
  const int node = blockIdx.x * 4 + wave;
  if (node >= NN) return;

  // LayerNorm over 128 (4 elems per lane)
  float loc[4];
  float sum = 0.f;
#pragma unroll
  for (int i = 0; i < 4; ++i) { loc[i] = hupd[(size_t)node * HID + lane + 32 * i]; sum += loc[i]; }
#pragma unroll
  for (int off = 16; off > 0; off >>= 1) sum += __shfl_xor(sum, off, 32);
  const float mu = sum * (1.f / 128.f);
  float vs = 0.f;
#pragma unroll
  for (int i = 0; i < 4; ++i) { float d = loc[i] - mu; vs += d * d; }
#pragma unroll
  for (int off = 16; off > 0; off >>= 1) vs += __shfl_xor(vs, off, 32);
  const float rstd = rsqrtf(vs * (1.f / 128.f) + 1e-5f);
#pragma unroll
  for (int i = 0; i < 4; ++i) {
    int o = lane + 32 * i;
    sH[wave][o] = (loc[i] - mu) * rstd * lng[o] + lnb[o];
  }
  // q = hln @ Wq^T
#pragma unroll
  for (int i = 0; i < 4; ++i) {
    int o = lane + 32 * i;
    float a = 0.f;
    for (int k = 0; k < 128; ++k) a += sH[wave][k] * Wq[o * 128 + k];
    sQ[wave][o] = a;
  }
  // scores [4 heads][64 tokens]
  const float scale = 0.17677669529663687f;  // 32^-0.5
#pragma unroll
  for (int t2 = 0; t2 < 2; ++t2) {
    int j = lane + 32 * t2;
#pragma unroll
    for (int hh = 0; hh < 4; ++hh) {
      float a = 0.f;
      for (int d = 0; d < 32; ++d) a += sQ[wave][hh * 32 + d] * kctx[j * 128 + hh * 32 + d];
      sS[wave][hh * 64 + j] = a * scale;
    }
  }
  // softmax per head (lanes 0..3 each own one head)
  if (lane < 4) {
    const int hh = lane;
    float mx = -3.4e38f;
    for (int j = 0; j < 64; ++j) mx = fmaxf(mx, sS[wave][hh * 64 + j]);
    float s = 0.f;
    for (int j = 0; j < 64; ++j) { float e = __expf(sS[wave][hh * 64 + j] - mx); sS[wave][hh * 64 + j] = e; s += e; }
    const float inv = 1.f / s;
    for (int j = 0; j < 64; ++j) sS[wave][hh * 64 + j] *= inv;
  }
  // out = attn @ V
#pragma unroll
  for (int i = 0; i < 4; ++i) {
    int o = lane + 32 * i;
    int hh = o >> 5;
    float a = 0.f;
    for (int j = 0; j < 64; ++j) a += sS[wave][hh * 64 + j] * vctx[j * 128 + o];
    sO[wave][o] = a;
  }
  // Wo projection + time influence + residual
  const int b = batch[node];
#pragma unroll
  for (int i = 0; i < 4; ++i) {
    int o = lane + 32 * i;
    float a = bo[o];
    for (int k = 0; k < 128; ++k) a += sO[wave][k] * Wo[o * 128 + k];
    float r = hio[(size_t)node * HID + o] + a + tinf[b * HID + o];
    hio[(size_t)node * HID + o] = r;
    hf16[(size_t)node * HID + o] = (_Float16)r;
  }
}

// ---------------------------------------------------------------------------
// Output heads: node_pred = mlp2(h, 128->256->12), pos_pred = mlp2(h, 128->128->3)
// ---------------------------------------------------------------------------
__global__ __launch_bounds__(128) void heads_kernel(
    const float* __restrict__ h,
    const float* __restrict__ npW1, const float* __restrict__ npb1,
    const float* __restrict__ npW2, const float* __restrict__ npb2,
    const float* __restrict__ ppW1, const float* __restrict__ ppb1,
    const float* __restrict__ ppW2, const float* __restrict__ ppb2,
    float* __restrict__ out) {
  __shared__ float sHid[4][256];
  const int wave = threadIdx.x >> 5, lane = threadIdx.x & 31;
  const int node = blockIdx.x * 4 + wave;
  if (node >= NN) return;
  const float* hrow = h + (size_t)node * HID;
  // node head hidden 256, silu
#pragma unroll
  for (int i = 0; i < 8; ++i) {
    int o = lane + 32 * i;
    float a = npb1[o];
    for (int k = 0; k < 128; ++k) a += hrow[k] * npW1[o * 128 + k];
    sHid[wave][o] = silu_f(a);
  }
  if (lane < 12) {
    float a = npb2[lane];
    for (int k = 0; k < 256; ++k) a += sHid[wave][k] * npW2[lane * 256 + k];
    out[(size_t)node * 12 + lane] = a;
  }
  // pos head hidden 128, silu (reuse sHid; same-wave DS ordering)
#pragma unroll
  for (int i = 0; i < 4; ++i) {
    int o = lane + 32 * i;
    float a = ppb1[o];
    for (int k = 0; k < 128; ++k) a += hrow[k] * ppW1[o * 128 + k];
    sHid[wave][o] = silu_f(a);
  }
  if (lane < 3) {
    float a = ppb2[lane];
    for (int k = 0; k < 128; ++k) a += sHid[wave][k] * ppW2[lane * 128 + k];
    out[(size_t)NN * 12 + (size_t)node * 3 + lane] = a;
  }
}

// ---------------------------------------------------------------------------
extern "C" void kernel_launch(void* const* d_in, const int* in_sizes, int n_in,
                              void* d_out, int out_size, void* d_ws, size_t ws_size,
                              hipStream_t stream) {
  (void)in_sizes; (void)n_in; (void)out_size; (void)ws_size;
  // inputs (setup_inputs dict insertion order; params dict flattened in order)
  const float* x    = (const float*)d_in[0];
  const int*   ei   = (const int*)d_in[1];
  const float* ea   = (const float*)d_in[2];
  // d_in[3] = pos (unused by reference network output path)
  const float* t    = (const float*)d_in[4];
  const float* topo = (const float*)d_in[5];
  const float* stab = (const float*)d_in[6];
  const float* sust = (const float*)d_in[7];
  const int*   batch = (const int*)d_in[8];
#define P(i) ((const float*)d_in[i])
  const float *nodeW = P(9), *nodeb = P(10), *edgeW = P(11), *edgeb = P(12);
  const float *tW1 = P(13), *tb1 = P(14), *tW2 = P(15), *tb2 = P(16);
  const float *toW1 = P(17), *tob1 = P(18), *toW2 = P(19), *tob2 = P(20);
  const float *stW1 = P(21), *stb1 = P(22), *stW2 = P(23), *stb2 = P(24);
  const float *suW1 = P(25), *sub1 = P(26), *suW2 = P(27), *sub2 = P(28);
  const float *cW1 = P(29), *cb1 = P(30), *cW2 = P(31), *cb2 = P(32);
  const float *Wq = P(33), *Wk = P(34), *Wv = P(35), *Wo = P(36), *bo = P(37);
  const float *msgW1 = P(38), *msgb1 = P(39), *msgW2 = P(40), *msgb2 = P(41);
  const float *updW1 = P(42), *updb1 = P(43), *updW2 = P(44), *updb2 = P(45);
  const float *lng = P(46), *lnb = P(47);
  const float *npW1 = P(48), *npb1 = P(49), *npW2 = P(50), *npb2 = P(51);
  const float *ppW1 = P(52), *ppb1 = P(53), *ppW2 = P(54), *ppb2 = P(55);
#undef P

  // workspace carve (256B aligned)
  char* wsp = (char*)d_ws;
  auto carve = [&](size_t bytes) -> char* {
    char* p = wsp;
    wsp += (bytes + 255) & ~(size_t)255;
    return p;
  };
  float*     h    = (float*)carve((size_t)NN * HID * 4);
  _Float16*  hf16 = (_Float16*)carve((size_t)NN * HID * 2);
  float*     agg  = (float*)carve((size_t)NN * HID * 4);
  float*     hupd = (float*)carve((size_t)NN * HID * 4);
  _Float16*  ef16 = (_Float16*)carve((size_t)NE * HID * 2);
  float*     tinf = (float*)carve((size_t)NG * HID * 4);
  float*     kctx = (float*)carve((size_t)64 * HID * 4);
  float*     vctx = (float*)carve((size_t)64 * HID * 4);

  // 0) tiny per-graph math
  precompute_kernel<<<1, 256, 0, stream>>>(
      t, topo, stab, sust, tW1, tb1, tW2, tb2, toW1, tob1, toW2, tob2,
      stW1, stb1, stW2, stb2, suW1, sub1, suW2, sub2, cW1, cb1, cW2, cb2,
      Wk, Wv, tinf, kctx, vctx);

  // 1) encoders
  node_enc_kernel<<<(NN * HID + 255) / 256, 256, 0, stream>>>(x, nodeW, nodeb, h, hf16);
  edge_enc_kernel<<<(unsigned)(((long long)NE * HID + 255) / 256), 256, 0, stream>>>(ea, edgeW, edgeb, ef16);

  // 2) six message-passing layers
  const int edgeTiles = NE / 16;   // 50000
  const int nodeTiles = NN / 16;   // 3125
  const int edgeBlocks = 1536;
  const int nodeBlocks = 512;
  for (int L = 0; L < 6; ++L) {
    zero_kernel<<<(NN * HID + 255) / 256, 256, 0, stream>>>(agg, NN * HID);
    edge_msg_kernel<<<edgeBlocks, 128, 0, stream>>>(
        ei, hf16, ef16,
        msgW1 + (size_t)L * 128 * 384, msgb1 + (size_t)L * 128,
        msgW2 + (size_t)L * 128 * 128, msgb2 + (size_t)L * 128,
        agg, edgeTiles, edgeBlocks * 4);
    node_upd_kernel<<<nodeBlocks, 128, 0, stream>>>(
        hf16, agg,
        updW1 + (size_t)L * 128 * 256, updb1 + (size_t)L * 128,
        updW2 + (size_t)L * 128 * 128, updb2 + (size_t)L * 128,
        hupd, nodeTiles, nodeBlocks * 4);
    ln_attn_kernel<<<NN / 4, 128, 0, stream>>>(
        hupd, h, hf16, lng + (size_t)L * 128, lnb + (size_t)L * 128,
        Wq, Wo, bo, kctx, vctx, tinf, batch);
  }

  // 3) output heads -> d_out (node_pred [50000*12] then pos_pred [50000*3])
  heads_kernel<<<NN / 4, 128, 0, stream>>>(
      h, npW1, npb1, npW2, npb2, ppW1, ppb1, ppW2, ppb2, (float*)d_out);
}
